// MultiQueryAttention_62251255988348
// MI455X (gfx1250) — matrix operand order repaired and checked by reading, not compiled, not run on hardware
//
#include <hip/hip_runtime.h>
#include <hip/hip_bf16.h>
#include <math.h>
#include <stdint.h>

typedef __bf16 bf16_t;
typedef __attribute__((ext_vector_type(16))) __bf16 v16bf;
typedef __attribute__((ext_vector_type(8)))  __bf16 v8bf;
typedef __attribute__((ext_vector_type(8)))  float  v8f;

#define WMMA_BF16(a, b, c) \
  __builtin_amdgcn_wmma_f32_16x16x32_bf16(false, (a), false, (b), (short)0, (c), false, false)

// Problem constants (from reference)
constexpr int Bb = 2, Ss = 2048, Ee = 1024, Hh = 16, KVh = 4, Dh = 64;

// ---------------------------------------------------------------------------
// Optional CDNA5 async global->LDS path (guarded; falls back to load+ds_store)
// Signature learned from hipcc diagnostic: param1 = AS1 v4i*, param2 = AS3 v4i*
// ---------------------------------------------------------------------------
#if defined(__has_builtin)
#  if __has_builtin(__builtin_amdgcn_global_load_async_to_lds_b128)
#    define HAVE_ASYNC_LDS 1
#  else
#    define HAVE_ASYNC_LDS 0
#  endif
#else
#  define HAVE_ASYNC_LDS 0
#endif

#if HAVE_ASYNC_LDS
typedef int v4i __attribute__((vector_size(16)));
typedef __attribute__((address_space(1))) v4i* g_v4i_p;   // global
typedef __attribute__((address_space(3))) v4i* l_v4i_p;   // LDS (32-bit ptr)
#  if __has_builtin(__builtin_amdgcn_s_wait_asynccnt)
#    define WAIT_ASYNC(n) __builtin_amdgcn_s_wait_asynccnt(n)
#  else
#    define WAIT_ASYNC(n) asm volatile("s_wait_asynccnt %0" ::"i"(n) : "memory")
#  endif
#else
#  define WAIT_ASYNC(n)
#endif

// ---------------------------------------------------------------------------
// fp32 -> bf16 convert
// ---------------------------------------------------------------------------
__global__ void conv_f32_bf16(const float* __restrict__ in, bf16_t* __restrict__ out, int n) {
  int i = blockIdx.x * 256 + threadIdx.x;
  if (i < n) out[i] = (bf16_t)in[i];
}

// ---------------------------------------------------------------------------
// GEMM: C[M,N] = A[M,K] * W[N,K]^T  (A,W bf16 row-major, C = bf16 or f32)
// Block = 8 waves cooperating on a 128(M) x 64(N) tile. The shared 64x32 W
// panel is staged in LDS (double-buffered, padded to 40-half rows so the
// ds_load_b128 fragment reads are bank-conflict-free), filled by
// GLOBAL_LOAD_ASYNC_TO_LDS_B128 (ASYNCcnt double-buffer pipeline) when
// available. Each wave does a 16x64 slab: per K-step 2 global b128 (A) +
// 8 ds b128 (W) feed 4 v_wmma_f32_16x16x32_bf16.
// ---------------------------------------------------------------------------
constexpr int WROW = 40;  // padded LDS row pitch in bf16 (80B: conflict-free b128)

template <typename OutT>
__global__ void __launch_bounds__(256)
gemm_wmma(const bf16_t* __restrict__ A, const bf16_t* __restrict__ W,
          OutT* __restrict__ C, int M, int N, int K) {
  __shared__ bf16_t wtile[2][64 * WROW];

  const int tid  = threadIdx.x;
  const int lane = tid & 31;
  const int wid  = tid >> 5;
  const int nblk = N >> 6;
  const int m0   = (blockIdx.x / nblk) * 128 + wid * 16;
  const int n0   = (blockIdx.x % nblk) * 64;
  const int hi   = lane >> 4;   // 0 = lanes 0-15, 1 = lanes 16-31
  const int r16  = lane & 15;
  const int ca   = hi * 16;     // A-fragment K chunk base

  // cooperative W staging: thread t moves 16B; row = t/4, 8-half segment = t%4
  const int wrow = tid >> 2;
  const int wseg = (tid & 3) * 8;
  const bf16_t* wsrc = W + (size_t)(n0 + wrow) * K + wseg;
  bf16_t* wdst0 = &wtile[0][wrow * WROW + wseg];
  bf16_t* wdst1 = &wtile[1][wrow * WROW + wseg];

  auto issue = [&](int k0, int buf) {
    bf16_t* dst = buf ? wdst1 : wdst0;
#if HAVE_ASYNC_LDS
    __builtin_amdgcn_global_load_async_to_lds_b128(
        (g_v4i_p)(uintptr_t)(wsrc + k0),
        (l_v4i_p)(uint32_t)(uintptr_t)dst,
        0, 0);
#else
    *(v8bf*)dst = *(const v8bf*)(wsrc + k0);
#endif
  };

  v8f acc[4] = {};
  const bf16_t* arow = A + (size_t)(m0 + r16) * K;
  const int nsteps = K >> 5;

  issue(0, 0);
  for (int i = 0; i < nsteps; ++i) {
    const int k0 = i << 5;
    if (i + 1 < nsteps) {
      issue(k0 + 32, (i + 1) & 1);
      WAIT_ASYNC(1);   // oldest (current buffer) complete, next still in flight
    } else {
      WAIT_ASYNC(0);
    }
    __syncthreads();   // current buffer visible to all waves

    // A fragment (16x32): halves 0-7 -> K = ca..ca+7, halves 8-15 -> ca+8..ca+15
    v16bf a;
    {
      v8bf lo  = *(const v8bf*)(arow + k0 + ca);
      v8bf hi8 = *(const v8bf*)(arow + k0 + ca + 8);
#pragma unroll
      for (int j = 0; j < 8; ++j) { a[j] = lo[j]; a[j + 8] = hi8[j]; }
    }
    const bf16_t* wb = &wtile[i & 1][0];
#pragma unroll
    for (int nt = 0; nt < 4; ++nt) {
      // B fragment (32x16): lane = column n, 16 contiguous K at offset 16*hi
      const bf16_t* brow = wb + (nt * 16 + r16) * WROW + hi * 16;
      v8bf b0 = *(const v8bf*)(brow);
      v8bf b1 = *(const v8bf*)(brow + 8);
      v16bf bf;
#pragma unroll
      for (int j = 0; j < 8; ++j) { bf[j] = b0[j]; bf[j + 8] = b1[j]; }
      acc[nt] = WMMA_BF16(a, bf, acc[nt]);
    }
    __syncthreads();   // all reads done before the buffer is refilled
  }

  // C/D layout: lane = column (N), VGPR r = row M (r + 8 for high lanes)
#pragma unroll
  for (int nt = 0; nt < 4; ++nt)
#pragma unroll
    for (int r = 0; r < 8; ++r) {
      int m = m0 + r + hi * 8;
      int n = n0 + nt * 16 + r16;
      C[(size_t)m * N + n] = (OutT)acc[nt][r];
    }
}

// ---------------------------------------------------------------------------
// In-place RoPE on bf16 buffer laid out [B, S, nheads*64].
// One thread per (b, s, head, j) with j in [0,32): rotates pair (j, j+32).
// ---------------------------------------------------------------------------
__global__ void rope_inplace(bf16_t* __restrict__ q, int nheads) {
  int idx = blockIdx.x * 256 + threadIdx.x;
  int total = Bb * Ss * nheads * 32;
  if (idx >= total) return;
  int j  = idx & 31;
  int t  = idx >> 5;
  int hh = t % nheads; t /= nheads;
  int s  = t % Ss;
  int b  = t / Ss;
  float invf = powf(10000.0f, -(float)j * (1.0f / 32.0f));
  float ang  = (float)s * invf;
  float sv, cv;
  sincosf(ang, &sv, &cv);
  size_t base = ((size_t)(b * Ss + s) * nheads + hh) * 64;
  float q1 = (float)q[base + j];
  float q2 = (float)q[base + j + 32];
  q[base + j]      = (bf16_t)(q1 * cv - q2 * sv);
  q[base + j + 32] = (bf16_t)(q2 * cv + q1 * sv);
}

// ---------------------------------------------------------------------------
// V transpose: [B,S,KV*64] -> [B,KV,64,S]  (so P*V B-operand K-dim is contiguous)
// ---------------------------------------------------------------------------
__global__ void transpose_v(const bf16_t* __restrict__ vb, bf16_t* __restrict__ vt) {
  int idx = blockIdx.x * 256 + threadIdx.x;
  int total = Bb * Ss * KVh * Dh;
  if (idx >= total) return;
  int d  = idx & 63;
  int t  = idx >> 6;
  int kv = t % KVh; t /= KVh;
  int s  = t % Ss;
  int b  = t / Ss;
  vt[(((size_t)b * KVh + kv) * 64 + d) * Ss + s] =
      vb[(size_t)(b * Ss + s) * (KVh * 64) + kv * 64 + d];
}

// ---------------------------------------------------------------------------
// Flash attention (causal, GQA). One wave per 16-query tile of one (b,h).
// Streams 32-key blocks: 2x WMMA chains for Q*K^T, online softmax with
// 16-lane shuffle reductions, P staged through LDS (D-layout -> A-layout),
// 4x WMMA for P*V. Output written bf16 [B,S,H*64].
// ---------------------------------------------------------------------------
__global__ void __launch_bounds__(256)
attn_fwd(const bf16_t* __restrict__ qb, const bf16_t* __restrict__ kbf,
         const bf16_t* __restrict__ vt, bf16_t* __restrict__ ob) {
  __shared__ bf16_t plds[8][16 * 32];  // per-wave P staging (1 KB each)

  const int lane = threadIdx.x & 31;
  const int wid  = threadIdx.x >> 5;
  const int tile = blockIdx.x * 8 + wid;      // B*H*(S/16) = 4096 tiles
  if (tile >= Bb * Hh * (Ss / 16)) return;
  const int qt = tile & 127;                  // S/16 = 128
  const int bh = tile >> 7;
  const int h  = bh & 15;
  const int b  = bh >> 4;
  const int kv = h >> 2;                      // G = H/KV = 4
  const int hi  = lane >> 4;
  const int r16 = lane & 15;
  const int ca  = hi * 16;
  const int q0  = qt << 4;

  // Q fragments (A operand), reused for all key blocks. Two K-steps of 32 over D=64.
  const bf16_t* qrow = qb + (size_t)(b * Ss + q0 + r16) * Ee + h * Dh;
  v16bf qf[2];
#pragma unroll
  for (int ks = 0; ks < 2; ++ks) {
    v8bf lo  = *(const v8bf*)(qrow + ks * 32 + ca);
    v8bf hi8 = *(const v8bf*)(qrow + ks * 32 + ca + 8);
#pragma unroll
    for (int i = 0; i < 8; ++i) { qf[ks][i] = lo[i]; qf[ks][i + 8] = hi8[i]; }
  }

  float mrow[8], lrow[8];
#pragma unroll
  for (int r = 0; r < 8; ++r) { mrow[r] = -INFINITY; lrow[r] = 0.0f; }
  v8f oacc[4] = {};
  const float scale = 0.125f;  // D^-0.5

  bf16_t* pl = &plds[wid][0];
  const bf16_t* kbase = kbf + (size_t)b * Ss * (KVh * Dh) + kv * Dh;
  const bf16_t* vbase = vt + ((size_t)(b * KVh + kv) * Dh) * Ss;

  for (int k0 = 0; k0 <= q0; k0 += 32) {
    // ---- scores: S = Q(16x64) * K^T(64x32) -> two 16x16 f32 fragments
    v8f s0 = {}, s1 = {};
#pragma unroll
    for (int ks = 0; ks < 2; ++ks) {
      const bf16_t* kr0 = kbase + (size_t)(k0 + r16) * (KVh * Dh) + ks * 32 + hi * 16;
      v8bf a0 = *(const v8bf*)kr0;
      v8bf a1 = *(const v8bf*)(kr0 + 8);
      v16bf kf0;
#pragma unroll
      for (int i = 0; i < 8; ++i) { kf0[i] = a0[i]; kf0[i + 8] = a1[i]; }
      s0 = WMMA_BF16(qf[ks], kf0, s0);

      const bf16_t* kr1 = kbase + (size_t)(k0 + 16 + r16) * (KVh * Dh) + ks * 32 + hi * 16;
      v8bf b0 = *(const v8bf*)kr1;
      v8bf b1 = *(const v8bf*)(kr1 + 8);
      v16bf kf1;
#pragma unroll
      for (int i = 0; i < 8; ++i) { kf1[i] = b0[i]; kf1[i + 8] = b1[i]; }
      s1 = WMMA_BF16(qf[ks], kf1, s1);
    }

    // ---- scale + causal mask + per-row tile max
    float tm[8];
    const int t0 = k0 + r16, t1 = t0 + 16;
#pragma unroll
    for (int r = 0; r < 8; ++r) {
      int sq = q0 + r + hi * 8;
      float a = s0[r] * scale; if (t0 > sq) a = -INFINITY;
      float c = s1[r] * scale; if (t1 > sq) c = -INFINITY;
      s0[r] = a; s1[r] = c;
      tm[r] = fmaxf(a, c);
    }
#pragma unroll
    for (int off = 1; off < 16; off <<= 1)
#pragma unroll
      for (int r = 0; r < 8; ++r)
        tm[r] = fmaxf(tm[r], __shfl_xor(tm[r], off, 32));

    // ---- online softmax update
    float rs[8];
#pragma unroll
    for (int r = 0; r < 8; ++r) {
      float nm    = fmaxf(mrow[r], tm[r]);
      float alpha = __expf(mrow[r] - nm);
      float p0 = __expf(s0[r] - nm);
      float p1 = __expf(s1[r] - nm);
      s0[r] = p0; s1[r] = p1;
      rs[r] = p0 + p1;
      mrow[r] = nm;
      lrow[r] *= alpha;
      oacc[0][r] *= alpha; oacc[1][r] *= alpha;
      oacc[2][r] *= alpha; oacc[3][r] *= alpha;
    }
#pragma unroll
    for (int off = 1; off < 16; off <<= 1)
#pragma unroll
      for (int r = 0; r < 8; ++r)
        rs[r] += __shfl_xor(rs[r], off, 32);
#pragma unroll
    for (int r = 0; r < 8; ++r) lrow[r] += rs[r];

    // ---- P: D-layout -> A-layout via per-wave LDS (intra-wave, in-order DS)
#pragma unroll
    for (int r = 0; r < 8; ++r) {
      int m = r + hi * 8;
      pl[m * 32 + r16]      = (bf16_t)s0[r];
      pl[m * 32 + 16 + r16] = (bf16_t)s1[r];
    }
    asm volatile("s_wait_dscnt 0" ::: "memory");
    v16bf pf;
    {
      v8bf lo  = *(const v8bf*)(pl + r16 * 32 + ca);
      v8bf hi8 = *(const v8bf*)(pl + r16 * 32 + ca + 8);
#pragma unroll
      for (int i = 0; i < 8; ++i) { pf[i] = lo[i]; pf[i + 8] = hi8[i]; }
    }

    // ---- O += P(16x32) * V(32x64)
#pragma unroll
    for (int nt = 0; nt < 4; ++nt) {
      const bf16_t* vr = vbase + (size_t)(nt * 16 + r16) * Ss + k0 + hi * 16;
      v8bf v0 = *(const v8bf*)vr;
      v8bf v1 = *(const v8bf*)(vr + 8);
      v16bf vf;
#pragma unroll
      for (int i = 0; i < 8; ++i) { vf[i] = v0[i]; vf[i + 8] = v1[i]; }
      oacc[nt] = WMMA_BF16(pf, vf, oacc[nt]);
    }
  }

  // ---- epilogue: normalize and store bf16 [B,S,H*64]
#pragma unroll
  for (int r = 0; r < 8; ++r) {
    float inv = 1.0f / lrow[r];
    int m = r + hi * 8;
    size_t base = (size_t)(b * Ss + q0 + m) * Ee + h * Dh;
#pragma unroll
    for (int nt = 0; nt < 4; ++nt)
      ob[base + nt * 16 + r16] = (bf16_t)(oacc[nt][r] * inv);
  }
}

// ---------------------------------------------------------------------------
// Launcher
// ---------------------------------------------------------------------------
extern "C" void kernel_launch(void* const* d_in, const int* in_sizes, int n_in,
                              void* d_out, int out_size, void* d_ws, size_t ws_size,
                              hipStream_t stream) {
  const float* x  = (const float*)d_in[0];
  const float* wq = (const float*)d_in[1];
  const float* wk = (const float*)d_in[2];
  const float* wv = (const float*)d_in[3];
  const float* wo = (const float*)d_in[4];
  float* out = (float*)d_out;

  const int n_x  = Bb * Ss * Ee;            // 4194304
  const int n_wq = Ee * Ee;                 // 1048576
  const int n_wk = KVh * Dh * Ee;           // 262144
  const int M    = Bb * Ss;                 // 4096 tokens
  const int Ekv  = KVh * Dh;                // 256

  // workspace layout (bytes)
  char* ws = (char*)d_ws;
  bf16_t* xb  = (bf16_t*)(ws);                       // 8 MB
  bf16_t* wqb = (bf16_t*)(ws + (8u << 20));          // 2 MB
  bf16_t* wkb = (bf16_t*)(ws + (10u << 20));         // 0.5 MB
  bf16_t* wvb = (bf16_t*)(ws + (10u << 20) + (512u << 10));
  bf16_t* wob = (bf16_t*)(ws + (11u << 20));         // 2 MB
  bf16_t* qb  = (bf16_t*)(ws + (13u << 20));         // 8 MB  [B,S,E]
  bf16_t* kb  = (bf16_t*)(ws + (21u << 20));         // 2 MB  [B,S,256]
  bf16_t* vb  = (bf16_t*)(ws + (23u << 20));         // 2 MB  [B,S,256]
  bf16_t* vtb = (bf16_t*)(ws + (25u << 20));         // 2 MB  [B,KV,64,S]
  bf16_t* ob  = (bf16_t*)(ws + (27u << 20));         // 8 MB  [B,S,E]

  // 1) convert inputs/weights to bf16
  conv_f32_bf16<<<(n_x  + 255) / 256, 256, 0, stream>>>(x,  xb,  n_x);
  conv_f32_bf16<<<(n_wq + 255) / 256, 256, 0, stream>>>(wq, wqb, n_wq);
  conv_f32_bf16<<<(n_wk + 255) / 256, 256, 0, stream>>>(wk, wkb, n_wk);
  conv_f32_bf16<<<(n_wk + 255) / 256, 256, 0, stream>>>(wv, wvb, n_wk);
  conv_f32_bf16<<<(n_wq + 255) / 256, 256, 0, stream>>>(wo, wob, n_wq);

  // 2) QKV projections (block-cooperative WMMA GEMM, bf16 out)
  {
    int blocks_q = (M / 128) * (Ee / 64);   // 512
    int blocks_k = (M / 128) * (Ekv / 64);  // 128
    gemm_wmma<bf16_t><<<blocks_q, 256, 0, stream>>>(xb, wqb, qb, M, Ee,  Ee);
    gemm_wmma<bf16_t><<<blocks_k, 256, 0, stream>>>(xb, wkb, kb, M, Ekv, Ee);
    gemm_wmma<bf16_t><<<blocks_k, 256, 0, stream>>>(xb, wvb, vb, M, Ekv, Ee);
  }

  // 3) RoPE in place on q and k
  rope_inplace<<<(Bb * Ss * Hh  * 32) / 256, 256, 0, stream>>>(qb, Hh);
  rope_inplace<<<(Bb * Ss * KVh * 32) / 256, 256, 0, stream>>>(kb, KVh);

  // 4) transpose V for the P*V operand layout
  transpose_v<<<(Bb * Ss * Ekv) / 256, 256, 0, stream>>>(vb, vtb);

  // 5) flash attention (causal, GQA)
  attn_fwd<<<(Bb * Hh * (Ss / 16)) / 8, 256, 0, stream>>>(qb, kb, vtb, ob);

  // 6) output projection -> f32 d_out
  {
    int blocks_o = (M / 128) * (Ee / 64);   // 512
    gemm_wmma<float><<<blocks_o, 256, 0, stream>>>(ob, wob, out, M, Ee, Ee);
  }
}
